// FeatDecoder_31756988186938
// MI455X (gfx1250) — compile-verified, matrix operand order checked by reading, not run on hardware
//
#include <hip/hip_runtime.h>
#include <hip/hip_bf16.h>

typedef __attribute__((ext_vector_type(16))) __bf16 v16bf;
typedef __attribute__((ext_vector_type(8)))  float  v8f;

union FragBF { v16bf v; uint4 q[2]; };
struct __align__(8) US4 { unsigned short x, y, z, w; };

#if defined(__gfx1250__)
#define HAVE_ASYNC 1
#else
#define HAVE_ASYNC 0
#endif

// 16-byte global -> LDS copy (CDNA5 async DMA path, ASYNCcnt-tracked).
// VDST operand = LDS byte offset (low 32 bits of generic shared pointer).
__device__ __forceinline__ void async_ld16(const void* g, void* l) {
#if HAVE_ASYNC
    unsigned lo = (unsigned)(unsigned long long)l;
    asm volatile("global_load_async_to_lds_b128 %0, %1, off"
                 :: "v"(lo), "v"(g) : "memory");
#else
    *(uint4*)l = *(const uint4*)g;
#endif
}
// 16-byte LDS -> global copy (async store path)
__device__ __forceinline__ void async_st16(void* g, const void* l) {
#if HAVE_ASYNC
    unsigned lo = (unsigned)(unsigned long long)l;
    asm volatile("global_store_async_from_lds_b128 %0, %1, off"
                 :: "v"(g), "v"(lo) : "memory");
#else
    *(uint4*)g = *(const uint4*)l;
#endif
}
__device__ __forceinline__ void wait_async() {
#if HAVE_ASYNC
#if __has_builtin(__builtin_amdgcn_s_wait_asynccnt)
    __builtin_amdgcn_s_wait_asynccnt(0);
#else
    asm volatile("s_wait_asynccnt 0x0" ::: "memory");
#endif
#endif
}

// hardware cvt (v_cvt_pk_bf16_f32) instead of bit-twiddled rounding
__device__ __forceinline__ unsigned short f2bf(float f) {
    __bf16 b = (__bf16)f;
    return __builtin_bit_cast(unsigned short, b);
}
__device__ __forceinline__ float bf2f(unsigned short s) {
    return __builtin_bit_cast(float, (unsigned)s << 16);
}
__device__ __forceinline__ void store4bf(unsigned short* dst, float4 v) {
    US4 p; p.x = f2bf(v.x); p.y = f2bf(v.y); p.z = f2bf(v.z); p.w = f2bf(v.w);
    *(US4*)dst = p;
}
__device__ __forceinline__ v16bf load_frag(const unsigned short* p) {
    FragBF f;
    f.q[0] = *(const uint4*)(p);        // k .. k+7
    f.q[1] = *(const uint4*)(p + 16);   // k+16 .. k+23
    return f.v;
}

// GEMM stage: D[TM,C] = epi(A[TM, KCH*C] @ W[KCH*C, C]).
// A in LDS bf16 (stride Apad). Weights pre-transposed bf16 [n][K] in global,
// staged to LDS [n][Kpc] via async 16B copies, in C-sized K chunks.
// MODE: 0 = relu(g*x+b) ; 1 = g*x+b ; 2 = relu(g*x+b + skip)   (all -> LDS)
template<int C, int TM, int KCH, int MODE>
__device__ void gemm_stage(const unsigned short* __restrict__ sAbuf, int Apad,
                           unsigned short* __restrict__ sW,
                           const unsigned short* __restrict__ wT,
                           const float* __restrict__ gvec,
                           const float* __restrict__ bvec,
                           unsigned short* dstL, int dstPad,
                           const unsigned short* skipL, int skipPad, int tid) {
    constexpr int Kpc = C + 8;
    constexpr int K   = KCH * C;
    constexpr int NT_M = TM / 16, NT_N = C / 16;
    constexpr int NTILES = NT_M * NT_N;
    constexpr int TPW = (NTILES + 7) / 8;       // 8 wave32 waves
    const int lane = tid & 31, wave = tid >> 5;
    const int lr = lane & 15, lh = lane >> 4;

    v8f acc[TPW];
    v8f vz = {};
    #pragma unroll
    for (int t = 0; t < TPW; ++t) acc[t] = vz;

    for (int ch = 0; ch < KCH; ++ch) {
        // stage W chunk: pure 16B async copies (8 bf16 each)
        for (int idx = tid; idx < C * (C / 8); idx += 256) {
            int n  = idx / (C / 8);
            int kc = (idx % (C / 8)) * 8;
            async_ld16(wT + (size_t)n * K + ch * C + kc, sW + n * Kpc + kc);
        }
        wait_async();
        __syncthreads();
        #pragma unroll
        for (int t = 0; t < TPW; ++t) {
            const int tile = wave + t * 8;
            if (tile < NTILES) {
                const int mt = tile / NT_N, nt = tile % NT_N;
                #pragma unroll
                for (int kk = 0; kk < C; kk += 32) {
                    v16bf a = load_frag(sAbuf + (mt * 16 + lr) * Apad + ch * C + kk + lh * 8);
                    v16bf b = load_frag(sW    + (nt * 16 + lr) * Kpc  +          kk + lh * 8);
                    acc[t] = __builtin_amdgcn_wmma_f32_16x16x32_bf16(
                        false, a, false, b, (short)0, acc[t], false, false);
                }
            }
        }
        __syncthreads();
    }
    #pragma unroll
    for (int t = 0; t < TPW; ++t) {
        const int tile = wave + t * 8;
        if (tile < NTILES) {
            const int mt = tile / NT_N, nt = tile % NT_N;
            const int n = nt * 16 + lr;
            const float G = gvec[n], B = bvec[n];
            #pragma unroll
            for (int v = 0; v < 8; ++v) {
                const int m = mt * 16 + lh * 8 + v;
                float val = acc[t][v] * G + B;
                if constexpr (MODE == 2) val += bf2f(skipL[m * skipPad + n]);
                if constexpr (MODE != 1) val = fmaxf(val, 0.f);
                dstL[m * dstPad + n] = f2bf(val);
            }
        }
    }
    __syncthreads();
}

// Fused decoder level. CF32: coarse input is f32 (level 2 reads x3),
// else coarse is bf16 (previous level's cur in workspace).
template<int C, int TM, bool CF32>
__global__ __launch_bounds__(256) void level_kernel(
    const void* __restrict__ coarse, const int* __restrict__ upmap,
    const float* __restrict__ xi, unsigned short* __restrict__ outCur, int N,
    const unsigned short* wUpT, const float* upg, const float* upb,
    const unsigned short* w1T, const float* g1, const float* b1,
    const unsigned short* w2T, const float* g2, const float* b2,
    const unsigned short* wsT, const float* gs, const float* bs,
    const unsigned short* v1T, const float* v1g, const float* v1b,
    const unsigned short* v2T, const float* v2g, const float* v2b) {
    constexpr int K2 = 2 * C, Kp2 = K2 + 8, Kpc = C + 8;
    __shared__ unsigned short sA[TM * Kp2];
    __shared__ unsigned short sB[TM * Kp2];
    __shared__ unsigned short sS[TM * Kpc];
    __shared__ unsigned short sW[C * Kpc];

    const int tid = threadIdx.x;
    const int r0  = blockIdx.x * TM;

    // ---- gather coarse rows -> sA
    if constexpr (CF32) {
        const float* co = (const float*)coarse;
        for (int idx = tid; idx < TM * (K2 / 4); idx += 256) {
            int t = idx / (K2 / 4), k0 = (idx % (K2 / 4)) * 4;
            int r = r0 + t;
            int src = (r < N) ? upmap[r] : 0;
            float4 v = *(const float4*)(co + (size_t)src * K2 + k0);
            store4bf(sA + t * Kp2 + k0, v);
        }
    } else {
        const unsigned short* co = (const unsigned short*)coarse;
        for (int idx = tid; idx < TM * (K2 / 8); idx += 256) {
            int t = idx / (K2 / 8), k0 = (idx % (K2 / 8)) * 8;
            int r = r0 + t;
            int src = (r < N) ? upmap[r] : 0;
            async_ld16(co + (size_t)src * K2 + k0, sA + t * Kp2 + k0);
        }
    }
    // ---- encoder skip xi (f32) -> sB[:, C:2C]
    for (int idx = tid; idx < TM * (C / 4); idx += 256) {
        int t = idx / (C / 4), k0 = (idx % (C / 4)) * 4;
        int r = r0 + t; if (r >= N) r = N - 1;
        float4 v = *(const float4*)(xi + (size_t)r * C + k0);
        store4bf(sB + t * Kp2 + C + k0, v);
    }
    wait_async();
    __syncthreads();

    // up = relu(bn(gather @ upW))   -> sB[:,0:C]  (cat = sB, width 2C)
    gemm_stage<C, TM, 2, 0>(sA, Kp2, sW, wUpT, upg, upb, sB, Kp2, nullptr, 0, tid);
    // s  = bn(cat @ Ws)             -> sS
    gemm_stage<C, TM, 2, 1>(sB, Kp2, sW, wsT, gs, bs, sS, Kpc, nullptr, 0, tid);
    // t1 = relu(bn(cat @ W1))       -> sA
    gemm_stage<C, TM, 2, 0>(sB, Kp2, sW, w1T, g1, b1, sA, Kpc, nullptr, 0, tid);
    // r1 = relu(bn(t1 @ W2) + s)    -> sB
    gemm_stage<C, TM, 1, 2>(sA, Kpc, sW, w2T, g2, b2, sB, Kpc, sS, Kpc, tid);
    // t3 = relu(bn(r1 @ V1))        -> sA
    gemm_stage<C, TM, 1, 0>(sB, Kpc, sW, v1T, v1g, v1b, sA, Kpc, nullptr, 0, tid);
    // out = relu(bn(t3 @ V2) + r1)  -> sS, then async-store to global bf16
    gemm_stage<C, TM, 1, 2>(sA, Kpc, sW, v2T, v2g, v2b, sS, Kpc, sB, Kpc, tid);

    for (int idx = tid; idx < TM * (C / 8); idx += 256) {
        int t = idx / (C / 8), k0 = (idx % (C / 8)) * 8;
        int r = r0 + t;
        if (r < N) async_st16(outCur + (size_t)r * C + k0, sS + t * Kpc + k0);
    }
    // S_ENDPGM performs implicit wait-idle; async stores drain before exit.
}

// One-time weight prep: f32 [K][C] -> bf16 transposed [C][K]
__global__ __launch_bounds__(256) void prep_w(const float* __restrict__ W,
                                              unsigned short* __restrict__ out,
                                              int K, int C) {
    int idx = blockIdx.x * 256 + threadIdx.x;
    if (idx >= K * C) return;
    int k = idx / C, n = idx % C;
    out[(size_t)n * K + k] = f2bf(W[idx]);
}

// Final 1x1 projection: out = [cur0(bf16), aux] @ projW + projb
__global__ __launch_bounds__(256) void proj_kernel(
    const unsigned short* __restrict__ cur0, const float* __restrict__ aux,
    const float* __restrict__ projW, const float* __restrict__ projb,
    float* __restrict__ out, int N) {
    int i = blockIdx.x * 256 + threadIdx.x;
    if (i >= N) return;
    const unsigned short* r = cur0 + (size_t)i * 32;
    float s = 0.f;
    #pragma unroll
    for (int c8 = 0; c8 < 4; ++c8) {
        uint4 q = *(const uint4*)(r + c8 * 8);
        unsigned w[4] = {q.x, q.y, q.z, q.w};
        #pragma unroll
        for (int j = 0; j < 4; ++j) {
            s += bf2f((unsigned short)(w[j] & 0xFFFFu)) * projW[c8 * 8 + 2 * j];
            s += bf2f((unsigned short)(w[j] >> 16))     * projW[c8 * 8 + 2 * j + 1];
        }
    }
    const float* a = aux + (size_t)i * 3;
    s += a[0] * projW[32] + a[1] * projW[33] + a[2] * projW[34];
    out[i] = s + projb[0];
}

extern "C" void kernel_launch(void* const* d_in, const int* in_sizes, int n_in,
                              void* d_out, int out_size, void* d_ws, size_t ws_size,
                              hipStream_t stream) {
    if (n_in < 64) return;
    // d_in: 0..3=x0..x3, 4..6=up_map0..2, 7=aux, 8..=param leaves per level (18):
    //   res1{W1,W2,Ws,b1,b2,bs,g1,g2,gs}, res2{W1,W2,b1,b2,g1,g2}, upW, upb, upg
    // 62=projW, 63=projb
    auto P = [&](int i) { return (const float*)d_in[i]; };

    unsigned short* cur0 = (unsigned short*)d_ws;            // 1e6   x 32 bf16
    unsigned short* cur1 = cur0 + (size_t)1000000 * 32;      // 250k  x 64
    unsigned short* cur2 = cur1 + (size_t)250000 * 64;       // 62.5k x 128
    unsigned short* wb   = cur2 + (size_t)62500 * 128;       // bf16 weights

    const int CHs[3] = {32, 64, 128};
    unsigned short* lvlw[3];
    {
        size_t off = 0;
        for (int l = 0; l < 3; ++l) { lvlw[l] = wb + off; off += (size_t)9 * CHs[l] * CHs[l]; }
    }
    const dim3 blk(256);

    // ---- prep: convert + transpose all weights to bf16 [n][K]
    for (int l = 0; l < 3; ++l) {
        const int C = CHs[l], L = 8 + l * 18;
        const size_t C2 = (size_t)C * C;
        unsigned short* base = lvlw[l];
        auto launch = [&](const float* W, unsigned short* o, int K) {
            int tot = K * C;
            prep_w<<<(tot + 255) / 256, blk, 0, stream>>>(W, o, K, C);
        };
        launch(P(L + 15), base + 0 * C2, 2 * C);   // upW_t
        launch(P(L + 0),  base + 2 * C2, 2 * C);   // res1.W1_t
        launch(P(L + 2),  base + 4 * C2, 2 * C);   // res1.Ws_t
        launch(P(L + 1),  base + 6 * C2, C);       // res1.W2_t
        launch(P(L + 9),  base + 7 * C2, C);       // res2.W1_t
        launch(P(L + 10), base + 8 * C2, C);       // res2.W2_t
    }

    auto run_level = [&](auto tagC, auto tagTM, auto tagCF32, const void* coarse,
                         const int* um, const float* xi, unsigned short* oc,
                         int N, int l) {
        constexpr int C  = decltype(tagC)::value;
        constexpr int TM = decltype(tagTM)::value;
        constexpr bool CF = decltype(tagCF32)::value;
        const int L = 8 + l * 18;
        const size_t C2 = (size_t)C * C;
        const unsigned short* base = lvlw[l];
        level_kernel<C, TM, CF><<<(N + TM - 1) / TM, blk, 0, stream>>>(
            coarse, um, xi, oc, N,
            base + 0 * C2, P(L + 17), P(L + 16),
            base + 2 * C2, P(L + 6),  P(L + 3),
            base + 6 * C2, P(L + 7),  P(L + 4),
            base + 4 * C2, P(L + 8),  P(L + 5),
            base + 7 * C2, P(L + 13), P(L + 11),
            base + 8 * C2, P(L + 14), P(L + 12));
    };
    using std::integral_constant;
    // level 2: c=128, coarse = x3 (f32)
    run_level(integral_constant<int, 128>{}, integral_constant<int, 16>{},
              integral_constant<bool, true>{},
              d_in[3], (const int*)d_in[6], (const float*)d_in[2], cur2, 62500, 2);
    // level 1: c=64, coarse = cur2 (bf16)
    run_level(integral_constant<int, 64>{}, integral_constant<int, 64>{},
              integral_constant<bool, false>{},
              cur2, (const int*)d_in[5], (const float*)d_in[1], cur1, 250000, 1);
    // level 0: c=32, coarse = cur1 (bf16)
    run_level(integral_constant<int, 32>{}, integral_constant<int, 64>{},
              integral_constant<bool, false>{},
              cur1, (const int*)d_in[4], (const float*)d_in[0], cur0, 1000000, 0);

    proj_kernel<<<(1000000 + 255) / 256, blk, 0, stream>>>(
        cur0, (const float*)d_in[7], P(62), P(63), (float*)d_out, 1000000);
}
